// OTAligner_40106404610847
// MI455X (gfx1250) — matrix operand order repaired and checked by reading, not compiled, not run on hardware
//
#include <hip/hip_runtime.h>

// ---------------------------------------------------------------------------
// Problem constants (from reference): B=16, I=512, J=2048, C_TEXT=512,
// C_MEL=80, ATTN_CH=80.
// ---------------------------------------------------------------------------
#define NB    16
#define NI    512
#define NJ    2048
#define NEGV  (-1000000000.0f)
#define TEMPV (0.0005f)
#define BLANKLP (-1.0f)

typedef __attribute__((ext_vector_type(16))) __bf16 v16bf;
typedef __attribute__((ext_vector_type(8)))  float  v8f;

static __device__ __forceinline__ __bf16 toBF(float x) { return (__bf16)x; }

union V16u { v16bf v; __bf16 h[16]; uint4 q[2]; };

// ------------------------------- reductions --------------------------------
static __device__ __forceinline__ float bmaxf(float v, float* red) {
  int tid = threadIdx.x;
  red[tid] = v; __syncthreads();
  for (int s = blockDim.x >> 1; s > 0; s >>= 1) {
    if (tid < s) red[tid] = fmaxf(red[tid], red[tid + s]);
    __syncthreads();
  }
  float r = red[0]; __syncthreads();
  return r;
}
static __device__ __forceinline__ float bsumf(float v, float* red) {
  int tid = threadIdx.x;
  red[tid] = v; __syncthreads();
  for (int s = blockDim.x >> 1; s > 0; s >>= 1) {
    if (tid < s) red[tid] = red[tid] + red[tid + s];
    __syncthreads();
  }
  float r = red[0]; __syncthreads();
  return r;
}
static __device__ __forceinline__ float lae(float a, float b) {
  float m = fmaxf(a, b), n = fminf(a, b);
  return m + log1pf(expf(n - m));
}

// ------------------------------ pack kernels -------------------------------
// A_k1[o][t*512+c] = k_w1[o][c][t]  (1024 x 1536 bf16)
__global__ void pack_kw1(const float* __restrict__ w, __bf16* __restrict__ A) {
  int idx = blockIdx.x * blockDim.x + threadIdx.x;
  if (idx >= 1024 * 1536) return;
  int o = idx / 1536, kk = idx % 1536;
  int t = kk / 512, c = kk % 512;
  A[idx] = toBF(w[(o * 512 + c) * 3 + t]);
}

// BT_k1[b][i][t*512+c] = text[b][i+t-1][c] (zero pad)  (B x 512 x 1536 bf16)
__global__ void pack_im2col_keys(const float* __restrict__ x, __bf16* __restrict__ BT) {
  size_t idx = (size_t)blockIdx.x * blockDim.x + threadIdx.x;
  if (idx >= (size_t)NB * NI * 1536) return;
  int kk = (int)(idx % 1536);
  size_t r = idx / 1536;
  int i = (int)(r % NI);
  int b = (int)(r / NI);
  int t = kk / 512, c = kk % 512;
  int ii = i + t - 1;
  float v = (ii >= 0 && ii < NI) ? x[((size_t)b * NI + ii) * 512 + c] : 0.0f;
  BT[idx] = toBF(v);
}

// A_q1[o][kk] (160 x 256 bf16), kk<240 -> t=kk/80, c=kk%80 -> q_w1[o][c][t]
__global__ void pack_qw1(const float* __restrict__ w, __bf16* __restrict__ A) {
  int idx = blockIdx.x * blockDim.x + threadIdx.x;
  if (idx >= 160 * 256) return;
  int o = idx / 256, kk = idx % 256;
  float v = 0.0f;
  if (kk < 240) { int t = kk / 80, c = kk % 80; v = w[(o * 80 + c) * 3 + t]; }
  A[idx] = toBF(v);
}

// BT_q1[b][j][kk] (B x 2048 x 256 bf16)
__global__ void pack_im2col_mel(const float* __restrict__ x, __bf16* __restrict__ BT) {
  size_t idx = (size_t)blockIdx.x * blockDim.x + threadIdx.x;
  if (idx >= (size_t)NB * NJ * 256) return;
  int kk = (int)(idx % 256);
  size_t r = idx / 256;
  int j = (int)(r % NJ);
  int b = (int)(r / NJ);
  float v = 0.0f;
  if (kk < 240) {
    int t = kk / 80, c = kk % 80;
    int jj = j + t - 1;
    if (jj >= 0 && jj < NJ) v = x[((size_t)b * NJ + jj) * 80 + c];
  }
  BT[idx] = toBF(v);
}

// Generic f32 -> bf16 copy with K padding (A[m][k<Kr ? w[m*Kr+k] : 0])
__global__ void pack_pad(const float* __restrict__ w, __bf16* __restrict__ A,
                         int M, int Kr, int Kp) {
  int idx = blockIdx.x * blockDim.x + threadIdx.x;
  if (idx >= M * Kp) return;
  int m = idx / Kp, k = idx % Kp;
  A[idx] = toBF(k < Kr ? w[m * Kr + k] : 0.0f);
}

// ------------------------------ WMMA GEMM ----------------------------------
// C(MxN) = A(MxK, row-major bf16) x B(KxN), with B given transposed (BT: NxK
// row-major bf16). f32 accumulate via v_wmma_f32_16x16x32_bf16.
// Each wave owns a 16(M) x 64(N) strip: one A fragment is reused by 4 WMMAs
// per K-step (4 accumulators), 4x better A reuse + 4x WMMA density.
// flags: bit0 = relu, bit1 = store f32 row-major (else store bf16 transposed
// Ct[n*ldc+m]).  M mult of 16; N mult of 256; K mult of 32.
// 128 threads = 4 waves -> block covers 16 x 256; grid.z = batch.
__global__ __launch_bounds__(128)
void gemm_bf16(const __bf16* __restrict__ A, const __bf16* __restrict__ BT,
               float* __restrict__ Cf, __bf16* __restrict__ Ct,
               const float* __restrict__ bias,
               int M, int N, int K, int lda, int ldbt, int ldc,
               size_t sA, size_t sBT, size_t sC, int flags) {
  int lane = threadIdx.x & 31;
  int wave = threadIdx.x >> 5;
  int b = blockIdx.z;
  int tileM = blockIdx.y * 16;
  int tileN = blockIdx.x * 256 + wave * 64;
  if (tileM >= M || tileN >= N) return;   // wave-uniform; EXEC stays all-ones
  int l15 = lane & 15;
  bool hi = lane >= 16;
  const __bf16* arow  = A + (size_t)b * sA + (size_t)(tileM + l15) * lda;
  const __bf16* brow0 = BT + (size_t)b * sBT + (size_t)(tileN + l15) * ldbt;
  const __bf16* brow1 = brow0 + (size_t)16 * ldbt;
  const __bf16* brow2 = brow0 + (size_t)32 * ldbt;
  const __bf16* brow3 = brow0 + (size_t)48 * ldbt;
  v8f acc0 = {}, acc1 = {}, acc2 = {}, acc3 = {};
  for (int k0 = 0; k0 < K; k0 += 32) {
    V16u av, bv0, bv1, bv2, bv3;
    // A fragment: lanes 0-15 K={0..7,16..23}, lanes 16-31 K={8..15,24..31}
    av.q[0] = *(const uint4*)(arow + k0 + (hi ? 8 : 0));
    av.q[1] = *(const uint4*)(arow + k0 + 16 + (hi ? 8 : 0));
    // B fragments: lanes 0-15 K={0..7,16..23}, lanes 16-31 K={8..15,24..31}
    int boff = k0 + (hi ? 8 : 0);
    bv0.q[0] = *(const uint4*)(brow0 + boff);
    bv0.q[1] = *(const uint4*)(brow0 + boff + 16);
    bv1.q[0] = *(const uint4*)(brow1 + boff);
    bv1.q[1] = *(const uint4*)(brow1 + boff + 16);
    bv2.q[0] = *(const uint4*)(brow2 + boff);
    bv2.q[1] = *(const uint4*)(brow2 + boff + 16);
    bv3.q[0] = *(const uint4*)(brow3 + boff);
    bv3.q[1] = *(const uint4*)(brow3 + boff + 16);
    acc0 = __builtin_amdgcn_wmma_f32_16x16x32_bf16(
        false, av.v, false, bv0.v, (short)0, acc0, false, false);
    acc1 = __builtin_amdgcn_wmma_f32_16x16x32_bf16(
        false, av.v, false, bv1.v, (short)0, acc1, false, false);
    acc2 = __builtin_amdgcn_wmma_f32_16x16x32_bf16(
        false, av.v, false, bv2.v, (short)0, acc2, false, false);
    acc3 = __builtin_amdgcn_wmma_f32_16x16x32_bf16(
        false, av.v, false, bv3.v, (short)0, acc3, false, false);
  }
  v8f accs[4] = {acc0, acc1, acc2, acc3};
#pragma unroll
  for (int s = 0; s < 4; ++s) {
    int n = tileN + s * 16 + l15;
#pragma unroll
    for (int r = 0; r < 8; ++r) {
      int m = tileM + r + (hi ? 8 : 0);
      float v = accs[s][r];
      if (bias) v += bias[m];
      if (flags & 1) v = fmaxf(v, 0.0f);
      if (flags & 2) Cf[(size_t)b * sC + (size_t)m * ldc + n] = v;
      else           Ct[(size_t)b * sC + (size_t)n * ldc + m] = toBF(v);
    }
  }
}

// ----------------------- sum of squares over 96 cols -----------------------
__global__ void row_sumsq(const __bf16* __restrict__ X, float* __restrict__ out,
                          int rows) {
  int r = blockIdx.x * blockDim.x + threadIdx.x;
  if (r >= rows) return;
  const __bf16* p = X + (size_t)r * 96;
  float s = 0.0f;
  for (int a = 0; a < 96; ++a) { float f = (float)p[a]; s += f * f; }
  out[r] = s;
}

// ------------- distance -> log_softmax -> +prior/mask -> softmax -----------
// One block per (b,j) row of length I=512; 256 threads, 2 elems each.
__global__ __launch_bounds__(256)
void attn_softmax(const float* __restrict__ S, const float* __restrict__ q2a,
                  const float* __restrict__ k2a, const float* __restrict__ prior,
                  const int* __restrict__ tmask,
                  float* __restrict__ logp_out, float* __restrict__ soft_out) {
  __shared__ float red[256];
  int row = blockIdx.x;            // b*NJ + j
  int b = row >> 11;               // NJ = 2048
  int tid = threadIdx.x;
  size_t base = (size_t)row * NI;
  float q2v = q2a[row];
  int i0 = tid, i1 = tid + 256;
  float lr0 = -TEMPV * (q2v + k2a[b * NI + i0] - 2.0f * S[base + i0]);
  float lr1 = -TEMPV * (q2v + k2a[b * NI + i1] - 2.0f * S[base + i1]);
  float mx = bmaxf(fmaxf(lr0, lr1), red);
  float sm = bsumf(expf(lr0 - mx) + expf(lr1 - mx), red);
  float lse = mx + logf(sm);
  float lp0 = lr0 - lse + logf(prior[base + i0] + 1e-8f);
  float lp1 = lr1 - lse + logf(prior[base + i1] + 1e-8f);
  if (tmask[b * NI + i0] <= 0) lp0 = NEGV;
  if (tmask[b * NI + i1] <= 0) lp1 = NEGV;
  logp_out[base + i0] = lp0;
  logp_out[base + i1] = lp1;
  float mx2 = bmaxf(fmaxf(lp0, lp1), red);
  float s2 = bsumf(expf(lp0 - mx2) + expf(lp1 - mx2), red);
  float inv = 1.0f / s2;
  soft_out[base + i0] = expf(lp0 - mx2) * inv;
  soft_out[base + i1] = expf(lp1 - mx2) * inv;
}

// ---------------------- Viterbi forward (per batch) ------------------------
__global__ __launch_bounds__(512)
void viterbi_fwd(const float* __restrict__ soft, const int* __restrict__ tmask,
                 const int* __restrict__ mmask, float* __restrict__ V) {
  __shared__ float vs[512];
  int b = blockIdx.x, x = threadIdx.x;
  bool tmx = tmask[b * NI + x] > 0;
  float v = NEGV;
  for (int y = 0; y < NJ; ++y) {
    vs[x] = v;
    __syncthreads();
    float vprev = (x == 0) ? ((y == 0) ? 0.0f : NEGV) : vs[x - 1];
    float vcur = (x == y) ? NEGV : v;
    bool mm = mmask[b * NJ + y] > 0;
    float col = (tmx && mm) ? soft[((size_t)b * NJ + y) * NI + x] : NEGV;
    float nv = fmaxf(vprev, vcur) + col;
    V[((size_t)b * NJ + y) * NI + x] = nv;
    v = nv;
    __syncthreads();
  }
}

// ---------------------- Viterbi backtrack (per batch) ----------------------
__global__ void viterbi_backtrack(const float* __restrict__ V,
                                  const int* __restrict__ tmask,
                                  const int* __restrict__ mmask,
                                  float* __restrict__ mas,
                                  float* __restrict__ dur) {
  int b = blockIdx.x;
  if (threadIdx.x != 0) return;
  int tx = 0;
  for (int i = 0; i < NI; ++i) tx += (tmask[b * NI + i] > 0) ? 1 : 0;
  int ty = 0;
  for (int j = 0; j < NJ; ++j) ty += (mmask[b * NJ + j] > 0) ? 1 : 0;
  int index = tx - 1;
  for (int y = NJ - 1; y >= 0; --y) {
    bool active = y < ty;
    if (active && index >= 0) {
      mas[((size_t)b * NJ + y) * NI + index] = 1.0f;
      dur[b * NI + index] += 1.0f;
    }
    float vcur = NEGV, vpre = NEGV;
    if (y > 0 && index >= 0) {
      const float* vp = &V[((size_t)b * NJ + (y - 1)) * NI];
      vcur = vp[index];
      vpre = vp[(index - 1 > 0) ? (index - 1) : 0];
    }
    bool move = (index != 0) && ((index == y) || (vcur < vpre)) && active;
    index -= move ? 1 : 0;
  }
}

// ---------------------- CTC forward-sum loss (per batch) -------------------
__global__ __launch_bounds__(512)
void ctc_fsl(const float* __restrict__ logp, const int* __restrict__ tmask,
             const int* __restrict__ mmask, float* __restrict__ part) {
  __shared__ float red[512];
  __shared__ float lp[513];
  __shared__ float albuf[2][1025];
  int b = blockIdx.x, tid = threadIdx.x;
  float kc = (tmask[b * NI + tid] > 0) ? 1.0f : 0.0f;
  int key_len = (int)(bsumf(kc, red) + 0.5f);
  float qc = 0.0f;
  for (int j = tid; j < NJ; j += 512) qc += (mmask[b * NJ + j] > 0) ? 1.0f : 0.0f;
  int query_len = (int)(bsumf(qc, red) + 0.5f);
  for (int s = tid; s < 1025; s += 512) albuf[0][s] = NEGV;
  __syncthreads();
  int cur = 0;
  int S2 = 2 * key_len + 1;
  for (int t = 0; t < NJ; ++t) {
    float lv = logp[((size_t)b * NJ + t) * NI + tid];
    lp[tid + 1] = (tid + 1 <= key_len) ? lv : NEGV;
    if (tid == 0) lp[0] = BLANKLP;
    __syncthreads();
    float pm = fmaxf(lp[tid], (tid == 0) ? lp[512] : -3.0e38f);
    float mx = bmaxf(pm, red);
    float ps = expf(lp[tid] - mx) + ((tid == 0) ? expf(lp[512] - mx) : 0.0f);
    float lse = mx + logf(bsumf(ps, red));
    int nxt = cur ^ 1;
#pragma unroll
    for (int ss = 0; ss < 3; ++ss) {
      int s = tid + ss * 512;
      if (s < 1025) {
        int lab = (s & 1) ? ((s + 1) >> 1) : 0;
        float e = lp[lab] - lse;
        float a1 = albuf[cur][s];
        float a2 = (s >= 1) ? albuf[cur][s - 1] : NEGV;
        float a3 = ((s >= 2) && (s & 1)) ? albuf[cur][s - 2] : NEGV;
        float prev = lae(lae(a1, a2), a3);
        if (t == 0) prev = (s <= 1) ? 0.0f : NEGV;
        float nv = (s < S2) ? prev + e : NEGV;
        albuf[nxt][s] = (t < query_len) ? nv : a1;
      }
    }
    __syncthreads();
    cur = nxt;
  }
  if (tid == 0) {
    float loss = 0.0f;
    if (key_len > 0) {
      float e1 = albuf[cur][2 * key_len];
      float e2 = albuf[cur][2 * key_len - 1];
      loss = -lae(e1, e2) / (float)key_len;
    }
    part[b] = loss;
  }
}

// ------------------------------- small utils -------------------------------
__global__ void copy_f32(const float* __restrict__ src, float* __restrict__ dst, int n) {
  int i = blockIdx.x * blockDim.x + threadIdx.x;
  if (i < n) dst[i] = src[i];
}
__global__ void loss_final(const float* __restrict__ part, float* __restrict__ out) {
  if (threadIdx.x == 0 && blockIdx.x == 0) {
    float s = 0.0f;
    for (int b = 0; b < NB; ++b) s += part[b];
    *out = s / (float)NB;
  }
}

// ------------------------------- launcher ----------------------------------
extern "C" void kernel_launch(void* const* d_in, const int* in_sizes, int n_in,
                              void* d_out, int out_size, void* d_ws, size_t ws_size,
                              hipStream_t stream) {
  (void)in_sizes; (void)n_in; (void)out_size; (void)ws_size;
  const float* text  = (const float*)d_in[0];
  const float* mel   = (const float*)d_in[1];
  const int*   tmask = (const int*)d_in[2];
  const int*   mmask = (const int*)d_in[3];
  const float* prior = (const float*)d_in[4];
  const float* k_w1  = (const float*)d_in[5];
  const float* k_b1  = (const float*)d_in[6];
  const float* k_w2  = (const float*)d_in[7];
  const float* k_b2  = (const float*)d_in[8];
  const float* q_w1  = (const float*)d_in[9];
  const float* q_b1  = (const float*)d_in[10];
  const float* q_w2  = (const float*)d_in[11];
  const float* q_b2  = (const float*)d_in[12];
  const float* q_w3  = (const float*)d_in[13];
  const float* q_b3  = (const float*)d_in[14];

  char* ws = (char*)d_ws;
  size_t off = 0;
  auto take = [&](size_t bytes) -> size_t {
    size_t o = (off + 255) & ~(size_t)255;
    off = o + bytes;
    return o;
  };
  const size_t B = NB;
  size_t o_Ak1  = take(1024 * 1536 * 2);
  size_t o_BTk1 = take(B * NI * 1536 * 2);
  size_t o_H1T  = take(B * NI * 1024 * 2);
  size_t o_Ak2  = take(80 * 1024 * 2);
  size_t o_kT   = take(B * NI * 96 * 2);
  size_t o_Aq1  = take(160 * 256 * 2);
  size_t o_BTq1 = take(B * NJ * 256 * 2);
  size_t o_H2T  = take(B * NJ * 160 * 2);
  size_t o_Aq2  = take(80 * 160 * 2);
  size_t o_BT5  = take(B * NJ * 96 * 2);
  size_t o_Aq3  = take(80 * 96 * 2);
  size_t o_qT   = take(B * NJ * 96 * 2);
  size_t o_k2   = take(B * NI * 4);
  size_t o_q2   = take(B * NJ * 4);
  size_t o_S    = take(B * NJ * NI * 4);   // reused as Viterbi V
  size_t o_dur  = take(B * NI * 4);
  size_t o_part = take(B * 4);

  __bf16* Ak1  = (__bf16*)(ws + o_Ak1);
  __bf16* BTk1 = (__bf16*)(ws + o_BTk1);
  __bf16* H1T  = (__bf16*)(ws + o_H1T);
  __bf16* Ak2  = (__bf16*)(ws + o_Ak2);
  __bf16* kT   = (__bf16*)(ws + o_kT);
  __bf16* Aq1  = (__bf16*)(ws + o_Aq1);
  __bf16* BTq1 = (__bf16*)(ws + o_BTq1);
  __bf16* H2T  = (__bf16*)(ws + o_H2T);
  __bf16* Aq2  = (__bf16*)(ws + o_Aq2);
  __bf16* BT5  = (__bf16*)(ws + o_BT5);
  __bf16* Aq3  = (__bf16*)(ws + o_Aq3);
  __bf16* qT   = (__bf16*)(ws + o_qT);
  float*  k2b  = (float*)(ws + o_k2);
  float*  q2b  = (float*)(ws + o_q2);
  float*  Sbuf = (float*)(ws + o_S);
  float*  durw = (float*)(ws + o_dur);
  float*  part = (float*)(ws + o_part);

  float* outp   = (float*)d_out;
  float* out_dur  = outp;                                   // B*NI
  float* out_soft = out_dur + (size_t)NB * NI;              // B*NJ*NI
  float* out_logp = out_soft + (size_t)NB * NJ * NI;        // B*NJ*NI
  float* out_mas  = out_logp + (size_t)NB * NJ * NI;        // B*NJ*NI
  float* out_loss = out_mas + (size_t)NB * NJ * NI;         // 1

  // Zero buffers whose K/M pads must stay zero, plus accumulation targets.
  hipMemsetAsync(kT, 0, B * NI * 96 * 2, stream);
  hipMemsetAsync(BT5, 0, B * NJ * 96 * 2, stream);
  hipMemsetAsync(qT, 0, B * NJ * 96 * 2, stream);
  hipMemsetAsync(durw, 0, B * NI * 4, stream);
  hipMemsetAsync(out_mas, 0, (size_t)NB * NJ * NI * 4, stream);

  // ---- packing ----
  pack_kw1<<<(1024 * 1536 + 255) / 256, 256, 0, stream>>>(k_w1, Ak1);
  pack_im2col_keys<<<(int)((B * NI * 1536 + 255) / 256), 256, 0, stream>>>(text, BTk1);
  pack_pad<<<(80 * 1024 + 255) / 256, 256, 0, stream>>>(k_w2, Ak2, 80, 1024, 1024);
  pack_qw1<<<(160 * 256 + 255) / 256, 256, 0, stream>>>(q_w1, Aq1);
  pack_im2col_mel<<<(int)((B * NJ * 256 + 255) / 256), 256, 0, stream>>>(mel, BTq1);
  pack_pad<<<(80 * 160 + 255) / 256, 256, 0, stream>>>(q_w2, Aq2, 80, 160, 160);
  pack_pad<<<(80 * 96 + 255) / 256, 256, 0, stream>>>(q_w3, Aq3, 80, 80, 96);

  dim3 blk(128);
  // G1: key conv  (M=1024,N=512,K=1536) -> H1T[b][i][o] (relu)
  gemm_bf16<<<dim3(512 / 256, 1024 / 16, NB), blk, 0, stream>>>(
      Ak1, BTk1, nullptr, H1T, k_b1, 1024, 512, 1536, 1536, 1536, 1024,
      0, (size_t)NI * 1536, (size_t)NI * 1024, /*relu*/ 1);
  // G2: key pw    (M=80,N=512,K=1024) -> kT[b][i][a]
  gemm_bf16<<<dim3(512 / 256, 80 / 16, NB), blk, 0, stream>>>(
      Ak2, H1T, nullptr, kT, k_b2, 80, 512, 1024, 1024, 1024, 96,
      0, (size_t)NI * 1024, (size_t)NI * 96, 0);
  // G3: mel conv  (M=160,N=2048,K=256) -> H2T[b][j][o] (relu)
  gemm_bf16<<<dim3(2048 / 256, 160 / 16, NB), blk, 0, stream>>>(
      Aq1, BTq1, nullptr, H2T, q_b1, 160, 2048, 256, 256, 256, 160,
      0, (size_t)NJ * 256, (size_t)NJ * 160, 1);
  // G4: mel pw1   (M=80,N=2048,K=160) -> BT5[b][j][a] (relu)
  gemm_bf16<<<dim3(2048 / 256, 80 / 16, NB), blk, 0, stream>>>(
      Aq2, H2T, nullptr, BT5, q_b2, 80, 2048, 160, 160, 160, 96,
      0, (size_t)NJ * 160, (size_t)NJ * 96, 1);
  // G5: mel pw2   (M=80,N=2048,K=96) -> qT[b][j][a]
  gemm_bf16<<<dim3(2048 / 256, 80 / 16, NB), blk, 0, stream>>>(
      Aq3, BT5, nullptr, qT, q_b3, 80, 2048, 96, 96, 96, 96,
      0, (size_t)NJ * 96, (size_t)NJ * 96, 0);
  // sum-of-squares
  row_sumsq<<<(NB * NI + 255) / 256, 256, 0, stream>>>(kT, k2b, NB * NI);
  row_sumsq<<<(NB * NJ + 255) / 256, 256, 0, stream>>>(qT, q2b, NB * NJ);
  // G6: scores    (M=2048,N=512,K=96) -> S[b][j][i] (f32)
  gemm_bf16<<<dim3(512 / 256, 2048 / 16, NB), blk, 0, stream>>>(
      qT, kT, Sbuf, nullptr, nullptr, 2048, 512, 96, 96, 96, 512,
      (size_t)NJ * 96, (size_t)NI * 96, (size_t)NJ * NI, /*f32*/ 2);

  // softmax rows -> logp, soft
  attn_softmax<<<NB * NJ, 256, 0, stream>>>(Sbuf, q2b, k2b, prior, tmask,
                                            out_logp, out_soft);
  // Viterbi (V reuses Sbuf)
  viterbi_fwd<<<NB, 512, 0, stream>>>(out_soft, tmask, mmask, Sbuf);
  viterbi_backtrack<<<NB, 1, 0, stream>>>(Sbuf, tmask, mmask, out_mas, durw);
  copy_f32<<<(NB * NI + 255) / 256, 256, 0, stream>>>(durw, out_dur, NB * NI);
  // CTC forward-sum loss
  ctc_fsl<<<NB, 512, 0, stream>>>(out_logp, tmask, mmask, part);
  loss_final<<<1, 1, 0, stream>>>(part, out_loss);
}
